// PointNet_FP_Module_11166914970522
// MI455X (gfx1250) — matrix operand order repaired and checked by reading, not compiled, not run on hardware
//
#include <hip/hip_runtime.h>

// ---- problem sizes (fixed by reference setup_inputs) ----
#define B_    4
#define N1_   8192
#define N2_   2048
#define C1_   128
#define C2_   256
#define CIN_  384       // C1 + C2
#define CO_   256       // mlp[0] == mlp[1]
#define M_    (B_ * N1_) // 32768 rows total

#define EPS_DIST 1e-07f
#define EPS_W    1e-08f
#define EPS_BN   1e-05f

typedef __attribute__((ext_vector_type(16))) __bf16 v16bf;
typedef __attribute__((ext_vector_type(8)))  float  v8f;

union Frag { v16bf v; unsigned int u[8]; };

__device__ __forceinline__ unsigned short f2bf(float f) {
  unsigned int u = __float_as_uint(f);
  u = (u + 0x7FFFu + ((u >> 16) & 1u)) >> 16;   // round-to-nearest-even
  return (unsigned short)u;
}
__device__ __forceinline__ unsigned int pack2bf(float a, float b) {
  return (unsigned int)f2bf(a) | ((unsigned int)f2bf(b) << 16);
}

// --- CDNA5 async global->LDS (ASYNCcnt path). ISA 10.2: low 32 bits of a
// generic LDS pointer are the wave-relative LDS byte address.
__device__ __forceinline__ void async_ld_b128(void* lds, const void* gaddr) {
  unsigned lds_off = (unsigned)(unsigned long long)lds;
  asm volatile("global_load_async_to_lds_b128 %0, %1, off"
               :: "v"(lds_off), "v"(gaddr)
               : "memory");
}
__device__ __forceinline__ void wait_async0() {
  asm volatile("s_wait_asynccnt 0x0" ::: "memory");
}

// ---------------------------------------------------------------------------
// K0: zero per-channel stats (sum,sumsq for both layers) -- ws is poisoned.
// ---------------------------------------------------------------------------
__global__ __launch_bounds__(256) void zero_stats_kernel(float* stats) {
  // stats layout: [sum0(256) | sq0(256) | sum1(256) | sq1(256)]
  for (int i = threadIdx.x; i < 4 * CO_; i += 256) stats[i] = 0.0f;
}

// ---------------------------------------------------------------------------
// K1: convert W0 (256x384) and W1 (256x256) f32 -> bf16 (row major N x K).
// ---------------------------------------------------------------------------
__global__ __launch_bounds__(256) void cvt_weights_kernel(
    const float* __restrict__ W0, const float* __restrict__ W1,
    unsigned short* __restrict__ w0b, unsigned short* __restrict__ w1b) {
  int idx = blockIdx.x * 256 + threadIdx.x;
  const int n0 = CO_ * CIN_;  // 98304
  const int n1 = CO_ * CO_;   // 65536
  if (idx < n0) w0b[idx] = f2bf(W0[idx]);
  else if (idx < n0 + n1) w1b[idx - n0] = f2bf(W1[idx - n0]);
}

// ---------------------------------------------------------------------------
// K2: three-interpolate + concat, output x0 (M x 384) in bf16.
//     grid = (N1/256, B), block = 256 (one point per thread)
// ---------------------------------------------------------------------------
__global__ __launch_bounds__(256) void interp_concat_kernel(
    const float* __restrict__ xyz1, const float* __restrict__ xyz2,
    const float* __restrict__ points1, const float* __restrict__ points2,
    unsigned short* __restrict__ x0) {
  __shared__ float s2[N2_ * 3];  // 24 KB: xyz2 for this batch
  const int b   = blockIdx.y;
  const int tid = threadIdx.x;
  for (int i = tid; i < N2_ * 3; i += 256) s2[i] = xyz2[(size_t)b * N2_ * 3 + i];
  __syncthreads();

  const int    p = blockIdx.x * 256 + tid;     // point index in N1
  const size_t m = (size_t)b * N1_ + p;        // global row
  const float ax = xyz1[m * 3 + 0];
  const float ay = xyz1[m * 3 + 1];
  const float az = xyz1[m * 3 + 2];

  float d0 = 1e30f, d1 = 1e30f, d2 = 1e30f;
  int   i0 = 0,     i1 = 0,     i2 = 0;
  #pragma unroll 4
  for (int n = 0; n < N2_; ++n) {
    float dx = ax - s2[3 * n + 0];
    float dy = ay - s2[3 * n + 1];
    float dz = az - s2[3 * n + 2];
    float d  = dx * dx + dy * dy + dz * dz;
    if (d < d0)      { d2 = d1; i2 = i1; d1 = d0; i1 = i0; d0 = d; i0 = n; }
    else if (d < d1) { d2 = d1; i2 = i1; d1 = d;  i1 = n; }
    else if (d < d2) { d2 = d;  i2 = n; }
  }
  // reference: clamp negative d2 to EPS_DIST (ours is >=0), sqrt, inv-dist weight
  float w0 = 1.0f / (sqrtf(d0 < 0.0f ? EPS_DIST : d0) + EPS_W);
  float w1 = 1.0f / (sqrtf(d1 < 0.0f ? EPS_DIST : d1) + EPS_W);
  float w2 = 1.0f / (sqrtf(d2 < 0.0f ? EPS_DIST : d2) + EPS_W);
  float ws = 1.0f / (w0 + w1 + w2);
  w0 *= ws; w1 *= ws; w2 *= ws;

  const float* p2 = points2 + (size_t)b * N2_ * C2_;
  const float4* r0 = (const float4*)(p2 + (size_t)i0 * C2_);
  const float4* r1 = (const float4*)(p2 + (size_t)i1 * C2_);
  const float4* r2 = (const float4*)(p2 + (size_t)i2 * C2_);
  unsigned int* dst = (unsigned int*)(x0 + m * CIN_);  // row is 16B aligned

  #pragma unroll 4
  for (int c4 = 0; c4 < C2_ / 4; ++c4) {
    float4 v0 = r0[c4], v1 = r1[c4], v2 = r2[c4];
    float o0 = w0 * v0.x + w1 * v1.x + w2 * v2.x;
    float o1 = w0 * v0.y + w1 * v1.y + w2 * v2.y;
    float o2 = w0 * v0.z + w1 * v1.z + w2 * v2.z;
    float o3 = w0 * v0.w + w1 * v1.w + w2 * v2.w;
    dst[c4 * 2 + 0] = pack2bf(o0, o1);
    dst[c4 * 2 + 1] = pack2bf(o2, o3);
  }
  const float4* p1 = (const float4*)(points1 + m * C1_);
  unsigned int* dst1 = dst + C2_ / 2;
  #pragma unroll 4
  for (int c4 = 0; c4 < C1_ / 4; ++c4) {
    float4 v = p1[c4];
    dst1[c4 * 2 + 0] = pack2bf(v.x, v.y);
    dst1[c4 * 2 + 1] = pack2bf(v.z, v.w);
  }
}

// ---------------------------------------------------------------------------
// K3: BF16 WMMA GEMM  Y[M x N] = A[M x K] * W[N x K]^T  (+ per-channel stats)
//     block tile 128(M) x 64(N), K-step 32, 8 waves of 32, wave tile 32x32
//     double-buffered LDS staged via GLOBAL_LOAD_ASYNC_TO_LDS_B128 (ASYNCcnt)
//     grid = (N/64, M/128)
// ---------------------------------------------------------------------------
__global__ __launch_bounds__(256) void gemm_bf16_stats_kernel(
    const unsigned short* __restrict__ A, const unsigned short* __restrict__ W,
    float* __restrict__ Y, float* __restrict__ stats, int M, int N, int K) {
  __shared__ unsigned short As[2][128][40];  // 32 elems + 8 pad (80B row, 16B aligned)
  __shared__ unsigned short Bs[2][64][40];   // total 30 KB

  const int tid  = threadIdx.x;
  const int wave = tid >> 5;
  const int lane = tid & 31;
  const int g    = lane >> 4;      // lane half
  const int l15  = lane & 15;
  const int m0   = blockIdx.y * 128;
  const int n0   = blockIdx.x * 64;
  const int wm   = (wave >> 1) * 32;  // 0,32,64,96
  const int wn   = (wave & 1) * 32;   // 0,32

  const int lrow = tid >> 2;          // 0..63
  const int lseg = (tid & 3) * 8;     // element offset 0,8,16,24

  const unsigned short* gA0 = A + (size_t)(m0 + lrow)      * K + lseg;
  const unsigned short* gA1 = A + (size_t)(m0 + lrow + 64) * K + lseg;
  const unsigned short* gB  = W + (size_t)(n0 + lrow)      * K + lseg;

  v8f acc[2][2];
  #pragma unroll
  for (int mi = 0; mi < 2; ++mi)
    #pragma unroll
    for (int ni = 0; ni < 2; ++ni)
      #pragma unroll
      for (int r = 0; r < 8; ++r) acc[mi][ni][r] = 0.0f;

  const int nk = K / 32;
  // prologue: stage tile 0 asynchronously into buffer 0
  async_ld_b128(&As[0][lrow][lseg],      gA0);
  async_ld_b128(&As[0][lrow + 64][lseg], gA1);
  async_ld_b128(&Bs[0][lrow][lseg],      gB);
  wait_async0();
  __syncthreads();

  for (int ks = 0; ks < nk; ++ks) {
    const int cur = ks & 1;
    const int k0  = ks * 32;
    if (ks + 1 < nk) {  // stage next tile into the other buffer while computing
      const int nxt = cur ^ 1;
      async_ld_b128(&As[nxt][lrow][lseg],      gA0 + k0 + 32);
      async_ld_b128(&As[nxt][lrow + 64][lseg], gA1 + k0 + 32);
      async_ld_b128(&Bs[nxt][lrow][lseg],      gB  + k0 + 32);
      if (ks + 2 < nk)  // hint the tile after that (global_prefetch_b8)
        __builtin_prefetch(gA0 + k0 + 64, 0, 1);
    }

    // A fragment: 16x32 bf16, ISA 7.12.2 layout
    Frag a[2], b[2];
    #pragma unroll
    for (int mi = 0; mi < 2; ++mi) {
      const int m = wm + mi * 16 + l15;
      #pragma unroll
      for (int v = 0; v < 8; ++v) {
        const int kb = 2 * v + ((v >= 4) ? 8 : 0) + 8 * g;  // even -> 32b LDS read
        a[mi].u[v] = *(const unsigned int*)&As[cur][m][kb];
      }
    }
    // B fragment: 32x16 bf16; same K order as A (lanes 0-15 K=0..7,16..23; lanes 16-31 K=8..15,24..31)
    #pragma unroll
    for (int ni = 0; ni < 2; ++ni) {
      const int n = wn + ni * 16 + l15;
      #pragma unroll
      for (int v = 0; v < 8; ++v) {
        const int kb = 2 * v + ((v >= 4) ? 8 : 0) + 8 * g;
        b[ni].u[v] = *(const unsigned int*)&Bs[cur][n][kb];
      }
    }
    #pragma unroll
    for (int mi = 0; mi < 2; ++mi)
      #pragma unroll
      for (int ni = 0; ni < 2; ++ni)
        acc[mi][ni] = __builtin_amdgcn_wmma_f32_16x16x32_bf16(
            false, a[mi].v, false, b[ni].v, (short)0, acc[mi][ni], false, false);

    wait_async0();     // next-tile LDS writes complete (this wave's)
    __syncthreads();   // all waves done reading cur + all writes visible
  }

  // epilogue: C/D layout -> VGPR r holds row (r + 8*lane_half), col = l15
  #pragma unroll
  for (int mi = 0; mi < 2; ++mi) {
    #pragma unroll
    for (int ni = 0; ni < 2; ++ni) {
      const int col   = n0 + wn + ni * 16 + l15;
      const int mbase = m0 + wm + mi * 16 + 8 * g;
      float s = 0.0f, s2 = 0.0f;
      #pragma unroll
      for (int r = 0; r < 8; ++r) {
        float val = acc[mi][ni][r];
        Y[(size_t)(mbase + r) * N + col] = val;
        s  += val;
        s2 += val * val;
      }
      atomicAdd(&stats[col], s);
      atomicAdd(&stats[N + col], s2);
    }
  }
}

// ---------------------------------------------------------------------------
// K4: BN params per channel: scale = gamma*rsqrt(var+eps), shift = beta - mean*scale
// ---------------------------------------------------------------------------
__global__ __launch_bounds__(256) void bn_params_kernel(
    const float* __restrict__ stats, const float* __restrict__ gamma,
    const float* __restrict__ beta, float* __restrict__ bnp, int count) {
  int c = threadIdx.x;  // 256 channels
  float inv  = 1.0f / (float)count;
  float mean = stats[c] * inv;
  float var  = stats[CO_ + c] * inv - mean * mean;  // biased, matches torch BN
  float scl  = gamma[c] * rsqrtf(var + EPS_BN);
  bnp[c]       = scl;
  bnp[CO_ + c] = beta[c] - mean * scl;
}

// ---------------------------------------------------------------------------
// K5: x1 = bf16(relu(bn0(y0)))   (4 consecutive channels per thread)
// ---------------------------------------------------------------------------
__global__ __launch_bounds__(256) void bn_relu_bf16_kernel(
    const float* __restrict__ y, const float* __restrict__ bnp,
    unsigned short* __restrict__ x1) {
  size_t idx  = (size_t)blockIdx.x * 256 + threadIdx.x;
  size_t base = idx * 4;
  if (base >= (size_t)M_ * CO_) return;
  int c = (int)(base & (CO_ - 1));
  float4 v = *(const float4*)(y + base);
  float o0 = fmaxf(v.x * bnp[c + 0] + bnp[CO_ + c + 0], 0.0f);
  float o1 = fmaxf(v.y * bnp[c + 1] + bnp[CO_ + c + 1], 0.0f);
  float o2 = fmaxf(v.z * bnp[c + 2] + bnp[CO_ + c + 2], 0.0f);
  float o3 = fmaxf(v.w * bnp[c + 3] + bnp[CO_ + c + 3], 0.0f);
  unsigned int* dst = (unsigned int*)(x1 + base);
  dst[0] = pack2bf(o0, o1);
  dst[1] = pack2bf(o2, o3);
}

// ---------------------------------------------------------------------------
// K6: out = relu(bn1(out))  in place on d_out (f32)
// ---------------------------------------------------------------------------
__global__ __launch_bounds__(256) void bn_relu_out_kernel(
    float* __restrict__ y, const float* __restrict__ bnp) {
  size_t idx  = (size_t)blockIdx.x * 256 + threadIdx.x;
  size_t base = idx * 4;
  if (base >= (size_t)M_ * CO_) return;
  int c = (int)(base & (CO_ - 1));
  float4 v = *(float4*)(y + base);
  v.x = fmaxf(v.x * bnp[c + 0] + bnp[CO_ + c + 0], 0.0f);
  v.y = fmaxf(v.y * bnp[c + 1] + bnp[CO_ + c + 1], 0.0f);
  v.z = fmaxf(v.z * bnp[c + 2] + bnp[CO_ + c + 2], 0.0f);
  v.w = fmaxf(v.w * bnp[c + 3] + bnp[CO_ + c + 3], 0.0f);
  *(float4*)(y + base) = v;
}

// ---------------------------------------------------------------------------
extern "C" void kernel_launch(void* const* d_in, const int* in_sizes, int n_in,
                              void* d_out, int out_size, void* d_ws, size_t ws_size,
                              hipStream_t stream) {
  const float* xyz1    = (const float*)d_in[0];
  const float* xyz2    = (const float*)d_in[1];
  const float* points1 = (const float*)d_in[2];
  const float* points2 = (const float*)d_in[3];
  const float* W0      = (const float*)d_in[4];
  const float* gamma0  = (const float*)d_in[5];
  const float* beta0   = (const float*)d_in[6];
  const float* W1      = (const float*)d_in[7];
  const float* gamma1  = (const float*)d_in[8];
  const float* beta1   = (const float*)d_in[9];
  float* out = (float*)d_out;

  // ---- workspace layout (bytes) ----
  char* ws = (char*)d_ws;
  const size_t X0_BYTES = (size_t)M_ * CIN_ * 2;  // 25,165,824
  const size_t Y0_BYTES = (size_t)M_ * CO_ * 4;   // 33,554,432
  unsigned short* x0  = (unsigned short*)(ws);
  unsigned short* x1  = x0;  // aliases x0 region (x0 dead after GEMM1)
  float*          y0  = (float*)(ws + X0_BYTES);
  unsigned short* w0b = (unsigned short*)(ws + X0_BYTES + Y0_BYTES);
  unsigned short* w1b = (unsigned short*)(ws + X0_BYTES + Y0_BYTES + (size_t)CO_ * CIN_ * 2);
  float* stats = (float*)(ws + X0_BYTES + Y0_BYTES + (size_t)CO_ * CIN_ * 2 + (size_t)CO_ * CO_ * 2);
  float* stats0 = stats;            // sum0 | sq0
  float* stats1 = stats + 2 * CO_;  // sum1 | sq1
  float* bnp0   = stats + 4 * CO_;  // scale0 | shift0
  float* bnp1   = stats + 6 * CO_;  // scale1 | shift1

  // K0: zero stats (ws is poisoned; must zero each launch)
  zero_stats_kernel<<<1, 256, 0, stream>>>(stats);

  // K1: weights -> bf16
  {
    int total = CO_ * CIN_ + CO_ * CO_;
    cvt_weights_kernel<<<(total + 255) / 256, 256, 0, stream>>>(W0, W1, w0b, w1b);
  }

  // K2: three-interpolate + concat -> x0 (bf16)
  interp_concat_kernel<<<dim3(N1_ / 256, B_), 256, 0, stream>>>(
      xyz1, xyz2, points1, points2, x0);

  // K3: GEMM1  y0 = x0 * W0^T   (M=32768, N=256, K=384) + stats0
  gemm_bf16_stats_kernel<<<dim3(CO_ / 64, M_ / 128), 256, 0, stream>>>(
      x0, w0b, y0, stats0, M_, CO_, CIN_);

  // K4: bn0 params
  bn_params_kernel<<<1, 256, 0, stream>>>(stats0, gamma0, beta0, bnp0, M_);

  // K5: x1 = bf16(relu(bn0(y0)))
  {
    size_t n4 = (size_t)M_ * CO_ / 4;
    bn_relu_bf16_kernel<<<(unsigned)((n4 + 255) / 256), 256, 0, stream>>>(y0, bnp0, x1);
  }

  // K3': GEMM2  out_raw = x1 * W1^T  (K=256) + stats1; raw y1 lands in d_out
  gemm_bf16_stats_kernel<<<dim3(CO_ / 64, M_ / 128), 256, 0, stream>>>(
      x1, w1b, out, stats1, M_, CO_, CO_);

  // K4': bn1 params
  bn_params_kernel<<<1, 256, 0, stream>>>(stats1, gamma1, beta1, bnp1, M_);

  // K6: out = relu(bn1(out)) in place
  {
    size_t n4 = (size_t)M_ * CO_ / 4;
    bn_relu_out_kernel<<<(unsigned)((n4 + 255) / 256), 256, 0, stream>>>(out, bnp1);
  }
}
